// Quantizer2_48558900249073
// MI455X (gfx1250) — matrix operand order repaired and checked by reading, not compiled, not run on hardware
//
#include <hip/hip_runtime.h>

// ---------------- problem constants (match reference) ----------------
#define ZD        256      // feature dim
#define NE        1000     // codebook entries
#define NEP       1024     // padded entries (64 WMMA n-tiles, 16 LDS chunks)
#define ROWS_WG   128      // rows per workgroup (8 waves * 16 rows)
#define CHUNK_E   64       // codebook entries staged in LDS per chunk
#define BROW      264      // LDS row pitch in bf16 elements (256 + 8 pad -> no bank conflicts)
#define LOSS_SCL  (0.3f / 256.0f)   // (0.1 + 0.2) * mean over D

typedef __attribute__((ext_vector_type(16))) __bf16    v16bf;
typedef __attribute__((ext_vector_type(8)))  float     v8f;
typedef __attribute__((ext_vector_type(8)))  unsigned  v8u;

struct alignas(16) F4 { float x, y, z, w; };
struct alignas(16) U4 { unsigned x, y, z, w; };
struct U4x2 { U4 a, b; };

// f32 -> bf16 round-to-nearest-even, packed pair (lo in [15:0], hi in [31:16])
__device__ __forceinline__ unsigned pk_bf16(float lo, float hi) {
    unsigned a = __float_as_uint(lo), b = __float_as_uint(hi);
    a = (a + 0x7FFFu + ((a >> 16) & 1u)) >> 16;
    b = (b + 0x7FFFu + ((b >> 16) & 1u)) >> 16;
    return a | (b << 16);
}

// ---------------- prep: codebook f32 -> bf16 (padded) + exact ||e||^2 ----------------
__global__ __launch_bounds__(256) void vq_prep(const float* __restrict__ emb,
                                               unsigned short* __restrict__ embB,
                                               float* __restrict__ e2g) {
    __shared__ float red[256];
    const int e = blockIdx.x;          // 0..NEP-1
    const int t = threadIdx.x;         // 0..255
    float x = (e < NE) ? emb[e * ZD + t] : 0.0f;
    unsigned u = __float_as_uint(x);
    u = (u + 0x7FFFu + ((u >> 16) & 1u)) >> 16;
    embB[e * ZD + t] = (unsigned short)u;
    red[t] = x * x;
    __syncthreads();
    #pragma unroll
    for (int s = 128; s > 0; s >>= 1) {
        if (t < s) red[t] += red[t + s];
        __syncthreads();
    }
    if (t == 0) e2g[e] = (e < NE) ? red[0] : 3.0e37f;   // sentinel: padded rows never win
}

// ---------------- main: WMMA distance + argmin + gather + loss ----------------
__global__ __launch_bounds__(256) void vq_main(const float* __restrict__ h,
                                               const float* __restrict__ emb,
                                               const unsigned short* __restrict__ embB,
                                               const float* __restrict__ e2g,
                                               float* __restrict__ outQ,
                                               int*   __restrict__ outIdx,
                                               float* __restrict__ outLoss) {
    __shared__ unsigned short ldsB[CHUNK_E * BROW];   // 33 KB staged bf16 codebook chunk
    __shared__ float sE2[CHUNK_E];
    __shared__ int   sIdx[ROWS_WG];

    const int tid  = threadIdx.x;
    const int lane = tid & 31;
    const int w    = tid >> 5;          // wave id 0..7
    const int half = lane >> 4;         // 0: lanes 0-15, 1: lanes 16-31
    const int l16  = lane & 15;

    const int waveRow = blockIdx.x * ROWS_WG + w * 16;
    const float* hrow = h + (long long)(waveRow + l16) * ZD;   // A-matrix: M = l16

    // ---- Build A fragments (bf16 of -2h), pinned in VGPRs; accumulate exact ||h||^2.
    // ISA 16-bit A 16x32 layout: lanes 0-15 hold K = kb+0..7 (u0..u3) and kb+16..23 (u4..u7),
    // lanes 16-31 hold K = kb+8..15 and kb+24..31.
    v16bf A[8];
    float hh = 0.0f;
    #pragma unroll
    for (int j = 0; j < 8; ++j) {
        const int kb = j * 32;
        F4 a0 = *(const F4*)(hrow + kb + half * 8);
        F4 a1 = *(const F4*)(hrow + kb + half * 8 + 4);
        F4 b0 = *(const F4*)(hrow + kb + 16 + half * 8);
        F4 b1 = *(const F4*)(hrow + kb + 16 + half * 8 + 4);
        hh += a0.x*a0.x + a0.y*a0.y + a0.z*a0.z + a0.w*a0.w;
        hh += a1.x*a1.x + a1.y*a1.y + a1.z*a1.z + a1.w*a1.w;
        hh += b0.x*b0.x + b0.y*b0.y + b0.z*b0.z + b0.w*b0.w;
        hh += b1.x*b1.x + b1.y*b1.y + b1.z*b1.z + b1.w*b1.w;
        v8u au;
        au[0] = pk_bf16(-2.0f*a0.x, -2.0f*a0.y);
        au[1] = pk_bf16(-2.0f*a0.z, -2.0f*a0.w);
        au[2] = pk_bf16(-2.0f*a1.x, -2.0f*a1.y);
        au[3] = pk_bf16(-2.0f*a1.z, -2.0f*a1.w);
        au[4] = pk_bf16(-2.0f*b0.x, -2.0f*b0.y);
        au[5] = pk_bf16(-2.0f*b0.z, -2.0f*b0.w);
        au[6] = pk_bf16(-2.0f*b1.x, -2.0f*b1.y);
        au[7] = pk_bf16(-2.0f*b1.z, -2.0f*b1.w);
        A[j] = __builtin_bit_cast(v16bf, au);
    }

    // ---- Running argmin state: lane holds column n = base + l16; c[v] is row M = v + 8*half.
    float bestS[8];
    int   bestN[8];
    #pragma unroll
    for (int v = 0; v < 8; ++v) { bestS[v] = 3.0e38f; bestN[v] = 0; }

    for (int chunk = 0; chunk < NEP / CHUNK_E; ++chunk) {
        __syncthreads();
        {   // cooperative stage: 64 entries x 256 bf16 -> LDS (padded rows)
            const int e = tid >> 2, part = tid & 3;
            const U4* src = (const U4*)(embB + ((chunk * CHUNK_E + e) * ZD + part * 64));
            U4* dst = (U4*)(ldsB + e * BROW + part * 64);
            #pragma unroll
            for (int q = 0; q < 8; ++q) dst[q] = src[q];
            if (tid < CHUNK_E) sE2[tid] = e2g[chunk * CHUNK_E + tid];
            if (chunk + 1 < NEP / CHUNK_E)   // prefetch next chunk into cache
                __builtin_prefetch(embB + (chunk + 1) * CHUNK_E * ZD + tid * 32, 0, 0);
        }
        __syncthreads();

        #pragma unroll
        for (int t = 0; t < CHUNK_E / 16; ++t) {
            v8f c = {};
            // B 32x16 layout: lane = column n (mod 16); lanes 0-15: K=kb+0..7,16..23, lanes 16-31: K=kb+8..15,24..31
            const unsigned short* bb = ldsB + (t * 16 + l16) * BROW + half * 8;
            #pragma unroll
            for (int j = 0; j < 8; ++j) {
                U4x2 braw;
                braw.a = ((const U4*)(bb + j * 32))[0];
                braw.b = ((const U4*)(bb + j * 32 + 16))[0];
                v16bf bv = __builtin_bit_cast(v16bf, braw);
                c = __builtin_amdgcn_wmma_f32_16x16x32_bf16(
                        false, A[j], false, bv, (short)0, c, false, false);
            }
            const int   n   = chunk * CHUNK_E + t * 16 + l16;
            const float e2n = sE2[t * 16 + l16];
            #pragma unroll
            for (int v = 0; v < 8; ++v) {
                const float s = e2n + c[v];      // = ||e||^2 - 2 h.e
                if (s < bestS[v]) { bestS[v] = s; bestN[v] = n; }
            }
        }
    }

    // ---- Finalize argmin across the 16 columns held in each half-wave (prefer smaller index on tie)
    #pragma unroll
    for (int m = 1; m <= 8; m <<= 1) {
        #pragma unroll
        for (int v = 0; v < 8; ++v) {
            const float oS = __shfl_xor(bestS[v], m);
            const int   oN = __shfl_xor(bestN[v], m);
            if (oS < bestS[v] || (oS == bestS[v] && oN < bestN[v])) {
                bestS[v] = oS; bestN[v] = oN;
            }
        }
    }
    hh += __shfl_xor(hh, 16);   // combine the two K-halves of the row's sum of squares

    // ---- Emit index + loss; stash indices for the gather phase
    #pragma unroll
    for (int v = 0; v < 8; ++v) {
        const int rloc = half * 8 + v;            // local row this (lane-half, v) reduced
        const float hhv = __shfl(hh, rloc);       // row rloc's ||h||^2 lives at lane rloc
        if (l16 == 0) {                           // lanes 0 and 16 are the reduced owners
            const int row = waveRow + rloc;
            sIdx[w * 16 + rloc] = bestN[v];
            outIdx[row]  = bestN[v];
            outLoss[row] = LOSS_SCL * (hhv + bestS[v]);   // 0.3 * mean((q-h)^2)
        }
    }
    __syncthreads();

    // ---- Gather quantized_st = emb[closest] (f32, L2-resident codebook)
    {
        const int idx = sIdx[w * 16 + l16];
        const F4* src = (const F4*)(emb + (long long)idx * ZD + half * 128);
        F4* dst = (F4*)(outQ + (long long)(waveRow + l16) * ZD + half * 128);
        #pragma unroll
        for (int q = 0; q < 32; ++q) dst[q] = src[q];
    }
}

// ---------------- launch ----------------
extern "C" void kernel_launch(void* const* d_in, const int* in_sizes, int n_in,
                              void* d_out, int out_size, void* d_ws, size_t ws_size,
                              hipStream_t stream) {
    (void)n_in; (void)out_size; (void)ws_size;
    const float* h   = (const float*)d_in[0];
    const float* emb = (const float*)d_in[1];
    const int Nrows  = in_sizes[0] / ZD;          // 131072

    unsigned short* embB = (unsigned short*)d_ws;           // NEP*ZD bf16 = 512 KB
    float*          e2g  = (float*)(embB + NEP * ZD);       // NEP f32

    float* outQ    = (float*)d_out;                                    // N*ZD f32
    int*   outIdx  = (int*)(outQ + (long long)Nrows * ZD);             // N int32
    float* outLoss = (float*)(outQ + (long long)Nrows * ZD + Nrows);   // N f32

    vq_prep<<<NEP, 256, 0, stream>>>(emb, embB, e2g);
    vq_main<<<Nrows / ROWS_WG, 256, 0, stream>>>(h, emb, embB, e2g,
                                                 outQ, outIdx, outLoss);
}
